// TransformerAttention_69011534512489
// MI455X (gfx1250) — matrix operand order repaired and checked by reading, not compiled, not run on hardware
//
#include <hip/hip_runtime.h>
#include <hip/hip_bf16.h>

#define D_MODEL 1024
#define NUM_HEADS 16
#define D_HEAD 64
#define BATCH 2
#define SEQ 2048
#define MTOT (BATCH * SEQ)

typedef __attribute__((ext_vector_type(16))) _Float16 v16h;
typedef __attribute__((ext_vector_type(8)))  _Float16 v8h;
typedef __attribute__((ext_vector_type(8)))  float    v8f;

static __device__ __forceinline__ v16h cat8(v8h lo, v8h hv) {
    v16h r;
#pragma unroll
    for (int j = 0; j < 8; ++j) { r[j] = lo[j]; r[j + 8] = hv[j]; }
    return r;
}

// A fragment 16x32 f16 (MxK) from row-major [*, ld] source.
// ISA layout: lane l holds row M=l%16; halves j<8 -> K=j+8*hi, j>=8 -> K=j+8+8*hi.
static __device__ __forceinline__ v16h load_a_frag(const _Float16* __restrict__ A,
                                                   int ld, int m0, int k0, int lane) {
    const int hi = lane >> 4;
    const _Float16* p = A + (size_t)(m0 + (lane & 15)) * ld + k0 + 8 * hi;
    v8h lo = *(const v8h*)p;          // K = k0+8*hi .. +7
    v8h hv = *(const v8h*)(p + 16);   // K = k0+16+8*hi .. +7
    return cat8(lo, hv);
}

// B fragment 32x16 f16 (KxN): source is N-major (row n holds contiguous K),
// ISA layout: lane l holds col N=l%16; halves j<8 -> K=j+8*hi, j>=8 -> K=j+8+8*hi.
// Works for both global and LDS pointers (row stride `ld` in halfs).
static __device__ __forceinline__ v16h load_b_frag(const _Float16* BT,
                                                   int ld, int n0, int k0, int lane) {
    const int hi = lane >> 4;
    const _Float16* p = BT + (size_t)(n0 + (lane & 15)) * ld + k0 + 8 * hi;
    v8h lo = *(const v8h*)p;
    v8h hv = *(const v8h*)(p + 16);
    return cat8(lo, hv);
}

static __device__ __forceinline__ v8f wmma_f16(v16h a, v16h b, v8f c) {
    return __builtin_amdgcn_wmma_f32_16x16x32_f16(false, a, false, b, (short)0, c,
                                                  false, false);
}

// CDNA5 async global->LDS copy: 32 bytes per call (two b128, ASYNCcnt += 2)
static __device__ __forceinline__ void async_copy32(uint32_t lds_addr, const _Float16* g) {
    uint64_t ga = (uint64_t)(size_t)g;
    asm volatile("global_load_async_to_lds_b128 %0, %1, off"
                 :: "v"(lds_addr), "v"(ga) : "memory");
    asm volatile("global_load_async_to_lds_b128 %0, %1, off offset:16"
                 :: "v"(lds_addr), "v"(ga) : "memory");
}

// ---------------- f32 -> f16 elementwise convert ----------------
__global__ __launch_bounds__(256) void cvt_f32_f16_kernel(const float* __restrict__ x,
                                                          _Float16* __restrict__ y, int n) {
    int i = (blockIdx.x * 256 + threadIdx.x) * 4;
    if (i + 3 < n) {
        float4 v = *(const float4*)(x + i);
        y[i + 0] = (_Float16)v.x;
        y[i + 1] = (_Float16)v.y;
        y[i + 2] = (_Float16)v.z;
        y[i + 3] = (_Float16)v.w;
    }
}

// ---------------- W[k][n] f32 -> WT[n][k] f16 (tiled LDS transpose) ----------------
__global__ __launch_bounds__(256) void transpose_w_kernel(const float* __restrict__ W,
                                                          _Float16* __restrict__ WT, int Dd) {
    __shared__ float tile[32][33];
    const int bx = blockIdx.x * 32;  // n
    const int by = blockIdx.y * 32;  // k
    const int tx = threadIdx.x & 31;
    const int ty = threadIdx.x >> 5;
#pragma unroll
    for (int r = ty; r < 32; r += 8)
        tile[r][tx] = W[(size_t)(by + r) * Dd + bx + tx];
    __syncthreads();
#pragma unroll
    for (int r = ty; r < 32; r += 8)
        WT[(size_t)(bx + r) * Dd + by + tx] = (_Float16)tile[tx][r];
}

// ---------------- GEMM: C = A[M,K] * BT[N,K]^T + bias, per-mode epilogue ----------------
// Shared B tile double-buffered through LDS with async global->LDS copies.
// mode 0: Q -> [B,H,S,Dh] f16 scaled; 1: K -> [B,H,S,Dh]; 2: V -> [B,H,Dh,S]; 3: f32 out.
__global__ __launch_bounds__(128) void gemm_bias_kernel(const _Float16* __restrict__ A,
                                                        const _Float16* __restrict__ BT,
                                                        const float* __restrict__ bias,
                                                        void* __restrict__ out,
                                                        int Kdim, int Ndim,
                                                        int mode, float scale) {
    __shared__ __align__(16) _Float16 ldsB[2][64][32];  // 2 x (64 n-rows x 32 k-halfs)
    const int tid  = threadIdx.x;
    const int lane = tid & 31;
    const int wave = tid >> 5;
    const int hi   = lane >> 4;
    const int m0   = blockIdx.y * 64 + wave * 16;
    const int n0   = blockIdx.x * 64;

    // staging: thread copies 32B of row (n0+sr): bytes [32*sc, 32*sc+32)
    const int sr = tid & 63;
    const int sc = tid >> 6;  // 0 or 1

    auto stage = [&](int buf, int k0) {
        async_copy32((uint32_t)(size_t)&ldsB[buf][sr][16 * sc],
                     BT + (size_t)(n0 + sr) * Kdim + k0 + 16 * sc);
    };

    v8f acc[4];
#pragma unroll
    for (int t = 0; t < 4; ++t)
#pragma unroll
        for (int j = 0; j < 8; ++j) acc[t][j] = 0.f;

    stage(0, 0);

#pragma unroll 1
    for (int k0 = 0; k0 < Kdim; k0 += 32) {
        const int buf = (k0 >> 5) & 1;
        if (k0 + 32 < Kdim) {
            stage(buf ^ 1, k0 + 32);  // prefetch next tile into the other buffer
            asm volatile("s_wait_asynccnt 0x2" ::: "memory");  // current tile landed
        } else {
            asm volatile("s_wait_asynccnt 0x0" ::: "memory");
        }
        __syncthreads();  // all waves' portions of this tile visible

        const _Float16* base = &ldsB[buf][0][0];
        v16h a  = load_a_frag(A, Kdim, m0, k0, lane);
        v16h b0 = load_b_frag(base, 32, 0,  0, lane);
        v16h b1 = load_b_frag(base, 32, 16, 0, lane);
        v16h b2 = load_b_frag(base, 32, 32, 0, lane);
        v16h b3 = load_b_frag(base, 32, 48, 0, lane);
        acc[0] = wmma_f16(a, b0, acc[0]);
        acc[1] = wmma_f16(a, b1, acc[1]);
        acc[2] = wmma_f16(a, b2, acc[2]);
        acc[3] = wmma_f16(a, b3, acc[3]);
        __syncthreads();  // reads done before this buffer is overwritten next iter
    }

    const int col = lane & 15;
#pragma unroll
    for (int t = 0; t < 4; ++t)
#pragma unroll
        for (int i = 0; i < 8; ++i) {
            const int m = m0 + i + 8 * hi;
            const int n = n0 + 16 * t + col;
            const float v = (acc[t][i] + bias[n]) * scale;
            if (mode == 3) {
                ((float*)out)[(size_t)m * Ndim + n] = v;
            } else {
                const int bb = m >> 11;          // m / SEQ
                const int s  = m & (SEQ - 1);
                const int hh = n >> 6;           // n / D_HEAD
                const int dh = n & (D_HEAD - 1);
                _Float16* o16 = (_Float16*)out;
                if (mode == 2)
                    o16[(((size_t)(bb * NUM_HEADS + hh)) * D_HEAD + dh) * SEQ + s] = (_Float16)v;
                else
                    o16[(((size_t)(bb * NUM_HEADS + hh)) * SEQ + s) * D_HEAD + dh] = (_Float16)v;
            }
        }
}

// ---------------- Flash attention: O = softmax(Q K^T) V ----------------
// All 4 waves of a block share (b,h): K/V tiles are staged ONCE per block into LDS
// via async global->LDS copies, double-buffered across the kv loop.
__global__ __launch_bounds__(128) void attn_kernel(const _Float16* __restrict__ Q,
                                                   const _Float16* __restrict__ K,
                                                   const _Float16* __restrict__ VT,
                                                   _Float16* __restrict__ O) {
    __shared__ __align__(16) _Float16 kT[2][32][64];    // kv rows x dh (8KB)
    __shared__ __align__(16) _Float16 vT[2][64][32];    // dh rows x kv (8KB)
    __shared__ __align__(16) _Float16 plds[4][16][48];  // per-wave P relayout (6KB)
    const int tid  = threadIdx.x;
    const int lane = tid & 31;
    const int wave = tid >> 5;
    const int hi   = lane >> 4;
    const int col  = lane & 15;
    const int bh   = blockIdx.y;               // b*H + h
    const int b    = bh / NUM_HEADS;
    const int h    = bh % NUM_HEADS;
    const int q0   = (blockIdx.x * 4 + wave) * 16;

    const _Float16* Qb = Q  + (size_t)bh * SEQ * D_HEAD;
    const _Float16* Kb = K  + (size_t)bh * SEQ * D_HEAD;
    const _Float16* Vb = VT + (size_t)bh * D_HEAD * SEQ;

    // staging assignments (32B per thread per tile)
    const int kr = tid >> 2, kc = tid & 3;   // K: 32 rows x 128B -> 4 chunks/row
    const int vr = tid & 63, vc = tid >> 6;  // V: 64 rows x 64B  -> 2 chunks/row

    auto stageKV = [&](int buf, int kv0) {
        async_copy32((uint32_t)(size_t)&kT[buf][kr][16 * kc],
                     Kb + (size_t)(kv0 + kr) * D_HEAD + 16 * kc);
        async_copy32((uint32_t)(size_t)&vT[buf][vr][16 * vc],
                     Vb + (size_t)vr * SEQ + kv0 + 16 * vc);
    };

    const v16h qa0 = load_a_frag(Qb, D_HEAD, q0, 0, lane);
    const v16h qa1 = load_a_frag(Qb, D_HEAD, q0, 32, lane);

    v8f oacc[4];
    float m_s[8], l_s[8];
#pragma unroll
    for (int t = 0; t < 4; ++t)
#pragma unroll
        for (int j = 0; j < 8; ++j) oacc[t][j] = 0.f;
#pragma unroll
    for (int i = 0; i < 8; ++i) { m_s[i] = -1.0e30f; l_s[i] = 0.f; }

    stageKV(0, 0);

#pragma unroll 1
    for (int kv0 = 0; kv0 < SEQ; kv0 += 32) {
        const int buf = (kv0 >> 5) & 1;
        if (kv0 + 32 < SEQ) {
            stageKV(buf ^ 1, kv0 + 32);
            asm volatile("s_wait_asynccnt 0x4" ::: "memory");  // current K+V landed
        } else {
            asm volatile("s_wait_asynccnt 0x0" ::: "memory");
        }
        __syncthreads();

        // ---- scores: two 16x16 tiles over 32 kv columns (in log2 domain) ----
        const _Float16* kbase = &kT[buf][0][0];
        v16h kb0 = load_b_frag(kbase, 64, 0,  0,  lane);
        v16h kb1 = load_b_frag(kbase, 64, 0,  32, lane);
        v16h kb2 = load_b_frag(kbase, 64, 16, 0,  lane);
        v16h kb3 = load_b_frag(kbase, 64, 16, 32, lane);
        v8f s0, s1;
#pragma unroll
        for (int j = 0; j < 8; ++j) { s0[j] = 0.f; s1[j] = 0.f; }
        s0 = wmma_f16(qa0, kb0, s0);
        s0 = wmma_f16(qa1, kb1, s0);
        s1 = wmma_f16(qa0, kb2, s1);
        s1 = wmma_f16(qa1, kb3, s1);

        // ---- online softmax: row reductions stay within 16-lane halves ----
        float cm[8];
#pragma unroll
        for (int i = 0; i < 8; ++i) cm[i] = fmaxf(s0[i], s1[i]);
#pragma unroll
        for (int off = 1; off < 16; off <<= 1)
#pragma unroll
            for (int i = 0; i < 8; ++i) cm[i] = fmaxf(cm[i], __shfl_xor(cm[i], off, 32));

        float corr[8], rs[8];
#pragma unroll
        for (int i = 0; i < 8; ++i) {
            const float nm = fmaxf(m_s[i], cm[i]);
            corr[i] = exp2f(m_s[i] - nm);
            m_s[i]  = nm;
            s0[i]   = exp2f(s0[i] - nm);
            s1[i]   = exp2f(s1[i] - nm);
            rs[i]   = s0[i] + s1[i];
        }
#pragma unroll
        for (int off = 1; off < 16; off <<= 1)
#pragma unroll
            for (int i = 0; i < 8; ++i) rs[i] += __shfl_xor(rs[i], off, 32);
#pragma unroll
        for (int i = 0; i < 8; ++i) l_s[i] = l_s[i] * corr[i] + rs[i];
#pragma unroll
        for (int t = 0; t < 4; ++t)
#pragma unroll
            for (int i = 0; i < 8; ++i) oacc[t][i] *= corr[i];

        // ---- C-layout probs -> A-layout via per-wave LDS (DS in-order per wave) ----
#pragma unroll
        for (int i = 0; i < 8; ++i) {
            plds[wave][i + 8 * hi][col]      = (_Float16)s0[i];
            plds[wave][i + 8 * hi][16 + col] = (_Float16)s1[i];
        }
        v16h pa;
        {
            const _Float16* r = &plds[wave][col][0];
            v8h lo = *(const v8h*)(r + 8 * hi);
            v8h hv = *(const v8h*)(r + 16 + 8 * hi);
            pa = cat8(lo, hv);
        }

        // ---- O += P @ V from the staged V tile ----
        const _Float16* vbase = &vT[buf][0][0];
        v16h vb0 = load_b_frag(vbase, 32, 0,  0, lane);
        v16h vb1 = load_b_frag(vbase, 32, 16, 0, lane);
        v16h vb2 = load_b_frag(vbase, 32, 32, 0, lane);
        v16h vb3 = load_b_frag(vbase, 32, 48, 0, lane);
        oacc[0] = wmma_f16(pa, vb0, oacc[0]);
        oacc[1] = wmma_f16(pa, vb1, oacc[1]);
        oacc[2] = wmma_f16(pa, vb2, oacc[2]);
        oacc[3] = wmma_f16(pa, vb3, oacc[3]);

        __syncthreads();  // tile reads done before next-iter staging overwrites
    }

    // ---- normalize, write O as [B,S,H*Dh] f16 so final projection is a plain GEMM ----
#pragma unroll
    for (int t = 0; t < 4; ++t)
#pragma unroll
        for (int i = 0; i < 8; ++i) {
            const int m = q0 + i + 8 * hi;
            const int n = 16 * t + col;
            const float v = oacc[t][i] / l_s[i];
            O[((size_t)(b * SEQ + m)) * D_MODEL + h * D_HEAD + n] = (_Float16)v;
        }
}

extern "C" void kernel_launch(void* const* d_in, const int* in_sizes, int n_in,
                              void* d_out, int out_size, void* d_ws, size_t ws_size,
                              hipStream_t stream) {
    (void)in_sizes; (void)n_in; (void)out_size; (void)ws_size;
    const float* seq = (const float*)d_in[0];
    const float* Wq  = (const float*)d_in[1];
    const float* bq  = (const float*)d_in[2];
    const float* Wk  = (const float*)d_in[3];
    const float* bk  = (const float*)d_in[4];
    const float* Wv  = (const float*)d_in[5];
    const float* bv  = (const float*)d_in[6];
    const float* Wo  = (const float*)d_in[7];
    const float* bo  = (const float*)d_in[8];

    _Float16* ws = (_Float16*)d_ws;
    size_t off = 0;
    _Float16* seq_h = ws + off; off += (size_t)MTOT * D_MODEL;
    _Float16* WqT   = ws + off; off += (size_t)D_MODEL * D_MODEL;
    _Float16* WkT   = ws + off; off += (size_t)D_MODEL * D_MODEL;
    _Float16* WvT   = ws + off; off += (size_t)D_MODEL * D_MODEL;
    _Float16* WoT   = ws + off; off += (size_t)D_MODEL * D_MODEL;
    _Float16* Qh    = ws + off; off += (size_t)MTOT * D_MODEL;  // [B,H,S,Dh]
    _Float16* Kh    = ws + off; off += (size_t)MTOT * D_MODEL;  // [B,H,S,Dh]
    _Float16* VTh   = ws + off; off += (size_t)MTOT * D_MODEL;  // [B,H,Dh,S]
    _Float16* Oh    = ws + off; off += (size_t)MTOT * D_MODEL;  // [B,S,D]

    // 1) convert activations to f16
    cvt_f32_f16_kernel<<<(MTOT * D_MODEL) / (256 * 4), 256, 0, stream>>>(
        seq, seq_h, MTOT * D_MODEL);

    // 2) transpose+convert all weights to N-major f16 (B-fragment friendly)
    dim3 tg(D_MODEL / 32, D_MODEL / 32);
    transpose_w_kernel<<<tg, 256, 0, stream>>>(Wq, WqT, D_MODEL);
    transpose_w_kernel<<<tg, 256, 0, stream>>>(Wk, WkT, D_MODEL);
    transpose_w_kernel<<<tg, 256, 0, stream>>>(Wv, WvT, D_MODEL);
    transpose_w_kernel<<<tg, 256, 0, stream>>>(Wo, WoT, D_MODEL);

    // 3) QKV projections (Q folds 1/sqrt(Dh) * log2e for exp2-domain softmax)
    dim3 gg(D_MODEL / 64, MTOT / 64);
    const float qscale = 0.125f * 1.44269504088896340736f;
    gemm_bias_kernel<<<gg, 128, 0, stream>>>(seq_h, WqT, bq, Qh,  D_MODEL, D_MODEL, 0, qscale);
    gemm_bias_kernel<<<gg, 128, 0, stream>>>(seq_h, WkT, bk, Kh,  D_MODEL, D_MODEL, 1, 1.0f);
    gemm_bias_kernel<<<gg, 128, 0, stream>>>(seq_h, WvT, bv, VTh, D_MODEL, D_MODEL, 2, 1.0f);

    // 4) flash attention
    dim3 ag(SEQ / 64, BATCH * NUM_HEADS);
    attn_kernel<<<ag, 128, 0, stream>>>(Qh, Kh, VTh, Oh);

    // 5) output projection -> f32 d_out
    gemm_bias_kernel<<<gg, 128, 0, stream>>>(Oh, WoT, bo, d_out, D_MODEL, D_MODEL, 3, 1.0f);
}
